// LearnableMultiheadSelfAttention_79620103733526
// MI455X (gfx1250) — matrix operand order repaired and checked by reading, not compiled, not run on hardware
//
#include <hip/hip_runtime.h>
#include <hip/hip_bf16.h>

typedef __bf16 bf16_t;
typedef bf16_t v8bf  __attribute__((ext_vector_type(8)));
typedef bf16_t v16bf __attribute__((ext_vector_type(16)));
typedef float  v8f   __attribute__((ext_vector_type(8)));
typedef unsigned int v4u __attribute__((ext_vector_type(4)));
typedef int  v4i __attribute__((ext_vector_type(4)));
typedef int  v8i __attribute__((ext_vector_type(8)));

#define WMMA_BF16(a, b, c) \
    __builtin_amdgcn_wmma_f32_16x16x32_bf16(false, (a), false, (b), (short)0, (c), false, false)

#if defined(__AMDGCN__) && __has_builtin(__builtin_amdgcn_tensor_load_to_lds) && \
    __has_builtin(__builtin_amdgcn_s_wait_tensorcnt)
#define USE_TDM 1
#else
#define USE_TDM 0
#endif

constexpr int L_  = 1024;
constexpr int B_  = 2;
constexpr int D_  = 1024;
constexpr int H_  = 16;
constexpr int DH_ = 64;
constexpr int M_  = 1024;
constexpr int T_  = L_ + M_;          // 2048
constexpr float SCALE_ = 0.125f;      // 1/sqrt(64)
constexpr float NEG_   = -1e30f;

// ---------------------------------------------------------------------------
// Elementwise helpers
// ---------------------------------------------------------------------------
__global__ void zero_f32(float* __restrict__ p, int n) {
    int i = blockIdx.x * blockDim.x + threadIdx.x;
    if (i < n) p[i] = 0.f;
}

__global__ void cvt_f32_bf16(const float* __restrict__ src, bf16_t* __restrict__ dst, int n) {
    int i = blockIdx.x * blockDim.x + threadIdx.x;
    if (i < n) dst[i] = (bf16_t)src[i];
}

// W[K][N] (row-major) -> WT[N][K] bf16 so B-fragments are contiguous rows.
__global__ void cvt_f32_bf16_T(const float* __restrict__ src, bf16_t* __restrict__ dst,
                               int K, int N) {
    int i = blockIdx.x * blockDim.x + threadIdx.x;
    if (i < K * N) {
        int k = i / N, n = i % N;
        dst[(size_t)n * K + k] = (bf16_t)src[i];
    }
}

// q_f32 [L*B, H*DH] + biases -> qu/qv bf16 [B][H][L][DH]
__global__ void repack_qu(const float* __restrict__ q, const float* __restrict__ u,
                          const float* __restrict__ v, bf16_t* __restrict__ qu,
                          bf16_t* __restrict__ qv) {
    int idx = blockIdx.x * blockDim.x + threadIdx.x;
    if (idx >= B_ * H_ * L_ * DH_) return;
    int d = idx % DH_; int t = idx / DH_;
    int i = t % L_;    t /= L_;
    int h = t % H_;    int b = t / H_;
    float base = q[((size_t)i * B_ + b) * (H_ * DH_) + h * DH_ + d];
    qu[idx] = (bf16_t)(base + u[h * DH_ + d]);
    qv[idx] = (bf16_t)(base + v[h * DH_ + d]);
}

// kv_f32 [T*B, 2*H*DH] -> k bf16 [B][H][T][DH],  vT bf16 [B][H][DH][T]
__global__ void repack_kv(const float* __restrict__ kv, bf16_t* __restrict__ kB,
                          bf16_t* __restrict__ vT) {
    int idx = blockIdx.x * blockDim.x + threadIdx.x;
    if (idx >= B_ * H_ * T_ * DH_) return;
    int d = idx % DH_; int t2 = idx / DH_;
    int t = t2 % T_;   t2 /= T_;
    int h = t2 % H_;   int b = t2 / H_;
    const size_t rowoff = ((size_t)t * B_ + b) * (2 * H_ * DH_);
    kB[idx] = (bf16_t)kv[rowoff + h * DH_ + d];
    float vv = kv[rowoff + H_ * DH_ + h * DH_ + d];
    vT[(((size_t)(b * H_ + h)) * DH_ + d) * T_ + t] = (bf16_t)vv;
}

// r_f32 [T*B, H*DH] -> r bf16 [B][H][T][DH]
__global__ void repack_r(const float* __restrict__ r, bf16_t* __restrict__ rB) {
    int idx = blockIdx.x * blockDim.x + threadIdx.x;
    if (idx >= B_ * H_ * T_ * DH_) return;
    int d = idx % DH_; int t2 = idx / DH_;
    int t = t2 % T_;   t2 /= T_;
    int h = t2 % H_;   int b = t2 / H_;
    rB[idx] = (bf16_t)r[((size_t)t * B_ + b) * (H_ * DH_) + h * DH_ + d];
}

// ---------------------------------------------------------------------------
// TDM descriptor helper: 2D bf16 tile (rows x rowElems) from a row-major
// tensor with pitch `pitchElems`, packed into LDS at byte offset `ldsAddr`.
// D# layout per CDNA5 ISA ch.8 (group0: count/lds/global/type=2;
// group1: data_size=1 (2B), tensor dims, tile dims, dim0 stride).
// 6-arg builtin (clang-23 lane): (v4u, v8i, v4i, v4i, v8i, i32 cpol).
// ---------------------------------------------------------------------------
#if USE_TDM
__device__ __forceinline__ void tdm_load_2d(const bf16_t* gptr, unsigned ldsAddr,
                                            unsigned rowElems, unsigned rows,
                                            unsigned pitchElems) {
    unsigned long long ga = (unsigned long long)(uintptr_t)gptr;
    v4u g0;
    g0.x = 1u;                                    // count=1, is_restore=0, gather off
    g0.y = ldsAddr;                               // lds_addr (bytes)
    g0.z = (unsigned)(ga & 0xFFFFFFFFu);          // global_addr[31:0]
    g0.w = (unsigned)((ga >> 32) & 0x01FFFFFFu) | (2u << 30);  // addr[56:32], type=2
    v8i g1;
    g1[0] = (int)(1u << 16);                      // workgroup_mask=0, data_size=1 (2B)
    g1[1] = (int)((rowElems & 0xFFFFu) << 16);    // tensor_dim0[15:0]
    g1[2] = (int)(((rowElems >> 16) & 0xFFFFu) | ((rows & 0xFFFFu) << 16)); // dim0 hi | dim1 lo
    g1[3] = (int)(((rows >> 16) & 0xFFFFu) | ((rowElems & 0xFFFFu) << 16)); // dim1 hi | tile_dim0
    g1[4] = (int)(rows & 0xFFFFu);                // tile_dim1, tile_dim2=0
    g1[5] = (int)pitchElems;                      // tensor_dim0_stride[31:0]
    g1[6] = 0;                                    // dim0_stride hi | dim1_stride lo
    g1[7] = 0;
    v4i zz4 = {};
    v8i zz8 = {};
    __builtin_amdgcn_tensor_load_to_lds(g0, g1, zz4, zz4, zz8, 0);
}
#endif

// ---------------------------------------------------------------------------
// WMMA "NT" GEMM: C[M,N] f32 = A[M,K] bf16 * BT[N,K]^T bf16.
// 128-thread block computes a 64x64 tile; 64x64 A/B panels staged in LDS via
// the Tensor Data Mover (wave 0 issues, s_wait_tensorcnt, then barrier).
// Wave w owns output rows [w*16, w*16+16).
// ---------------------------------------------------------------------------
__global__ __launch_bounds__(128) void gemm_nt_bf16(const bf16_t* __restrict__ A,
                                                    const bf16_t* __restrict__ BT,
                                                    float* __restrict__ C,
                                                    int Mdim, int Ndim, int K) {
    __shared__ bf16_t Apanel[64][64];
    __shared__ bf16_t Bpanel[64][64];

    const int lane = threadIdx.x & 31;
    const int wv   = threadIdx.x >> 5;    // wave id 0..3 (wave32)
    const int half = lane >> 4;
    const int l15  = lane & 15;
    const int row0 = blockIdx.x * 64;
    const int col0 = blockIdx.y * 64;

    const bf16_t* Ablk = A  + (size_t)row0 * K;
    const bf16_t* Bblk = BT + (size_t)col0 * K;

#if USE_TDM
    const unsigned ldsA = (unsigned)(uintptr_t)&Apanel[0][0];
    const unsigned ldsB = (unsigned)(uintptr_t)&Bpanel[0][0];
#endif

    v8f acc0 = {}, acc1 = {}, acc2 = {}, acc3 = {};

    for (int kc = 0; kc < K; kc += 64) {
        __syncthreads();   // previous chunk fully consumed
#if USE_TDM
        if (wv == 0) {     // wave-uniform: one TDM issue per panel
            tdm_load_2d(Ablk + kc, ldsA, 64u, 64u, (unsigned)K);
            tdm_load_2d(Bblk + kc, ldsB, 64u, 64u, (unsigned)K);
            __builtin_amdgcn_s_wait_tensorcnt(0);
        }
#else
        for (int t = threadIdx.x; t < 64 * 64 / 8; t += 128) {
            int r = t >> 3, c = (t & 7) * 8;
            *(v8bf*)&Apanel[r][c] = *(const v8bf*)(Ablk + (size_t)r * K + kc + c);
            *(v8bf*)&Bpanel[r][c] = *(const v8bf*)(Bblk + (size_t)r * K + kc + c);
        }
#endif
        __syncthreads();   // panels visible to all waves

#pragma unroll
        for (int ks = 0; ks < 2; ++ks) {
            // A fragment (16x32): lane = row within wave's 16 rows
            const bf16_t* ap = &Apanel[wv * 16 + l15][ks * 32 + half * 16];
            v8bf alo = *(const v8bf*)ap;
            v8bf ahi = *(const v8bf*)(ap + 8);
            v16bf a = __builtin_shufflevector(alo, ahi, 0, 1, 2, 3, 4, 5, 6, 7,
                                              8, 9, 10, 11, 12, 13, 14, 15);
            // B fragments (32x16 each): lane = column, contiguous K run
            v16bf b0 = *(const v16bf*)&Bpanel[0 * 16 + l15][ks * 32 + half * 16];
            v16bf b1 = *(const v16bf*)&Bpanel[1 * 16 + l15][ks * 32 + half * 16];
            v16bf b2 = *(const v16bf*)&Bpanel[2 * 16 + l15][ks * 32 + half * 16];
            v16bf b3 = *(const v16bf*)&Bpanel[3 * 16 + l15][ks * 32 + half * 16];
            acc0 = WMMA_BF16(a, b0, acc0);
            acc1 = WMMA_BF16(a, b1, acc1);
            acc2 = WMMA_BF16(a, b2, acc2);
            acc3 = WMMA_BF16(a, b3, acc3);
        }
    }

    float* Cr = C + (size_t)(row0 + wv * 16) * Ndim + col0;
#pragma unroll
    for (int r = 0; r < 8; ++r) {
        const size_t roff = (size_t)(r + half * 8) * Ndim;
        Cr[roff + 0  + l15] = acc0[r];
        Cr[roff + 16 + l15] = acc1[r];
        Cr[roff + 32 + l15] = acc2[r];
        Cr[roff + 48 + l15] = acc3[r];
    }
}

// ---------------------------------------------------------------------------
// Attention: 1 wave per (b, h, 16-row tile). Two-pass streaming softmax.
// ---------------------------------------------------------------------------
__global__ __launch_bounds__(32) void attn_kernel(const bf16_t* __restrict__ qu,
                                                  const bf16_t* __restrict__ qv,
                                                  const bf16_t* __restrict__ kk,
                                                  const bf16_t* __restrict__ vT,
                                                  const bf16_t* __restrict__ rr,
                                                  float* __restrict__ attnM,
                                                  bf16_t* __restrict__ avec) {
    __shared__ float bdlds[16][32];   // rel-shift staging band
    __shared__ float pbuf[16][32];    // probability tile -> A-fragment staging

    const int lane = threadIdx.x;
    const int half = lane >> 4;
    const int l15  = lane & 15;
    const int i0   = blockIdx.x * 16;
    const int bh   = blockIdx.y;
    const int b    = bh / H_;
    const int h    = bh % H_;

    const bf16_t* quB = qu + (size_t)bh * L_ * DH_;
    const bf16_t* qvB = qv + (size_t)bh * L_ * DH_;
    const bf16_t* kB  = kk + (size_t)bh * T_ * DH_;
    const bf16_t* rB  = rr + (size_t)bh * T_ * DH_;
    const bf16_t* vB  = vT + (size_t)bh * DH_ * T_;

    const int arow = i0 + l15;
    auto loadA = [&](const bf16_t* base, int k0) -> v16bf {
        const bf16_t* p = base + (size_t)arow * DH_ + k0 + half * 16;
        v8bf lo = *(const v8bf*)p;
        v8bf hi = *(const v8bf*)(p + 8);
        return __builtin_shufflevector(lo, hi, 0, 1, 2, 3, 4, 5, 6, 7,
                                       8, 9, 10, 11, 12, 13, 14, 15);
    };
    const v16bf au0 = loadA(quB, 0), au1 = loadA(quB, 32);
    const v16bf av0 = loadA(qvB, 0), av1 = loadA(qvB, 32);

    // Score tile (16 rows x 16 cols) at column offset j0: AC + rel-shifted BD.
    auto s_tile = [&](int j0) -> v8f {
        const int jc = j0 + l15;
        v8f ac = {};
        ac = WMMA_BF16(au0, *(const v16bf*)(kB + (size_t)jc * DH_ + 0  + half * 16), ac);
        ac = WMMA_BF16(au1, *(const v16bf*)(kB + (size_t)jc * DH_ + 32 + half * 16), ac);
        const int c0 = j0 + L_ - 16 - i0;   // raw-BD band start, >= 0 for all tiles
#pragma unroll
        for (int t = 0; t < 2; ++t) {
            int cc = c0 + t * 16 + l15;
            cc = cc > (T_ - 1) ? (T_ - 1) : cc;   // OOB only in masked region
            v8f bd = {};
            bd = WMMA_BF16(av0, *(const v16bf*)(rB + (size_t)cc * DH_ + 0  + half * 16), bd);
            bd = WMMA_BF16(av1, *(const v16bf*)(rB + (size_t)cc * DH_ + 32 + half * 16), bd);
#pragma unroll
            for (int r = 0; r < 8; ++r)
                bdlds[r + half * 8][t * 16 + l15] = bd[r];
        }
        __syncthreads();
        v8f s;
#pragma unroll
        for (int r = 0; r < 8; ++r) {
            const int m = r + half * 8;
            const int i = i0 + m;
            const int j = j0 + l15;
            float val = (ac[r] + bdlds[m][l15 + 15 - m]) * SCALE_;
            s[r] = (j > i + M_) ? NEG_ : val;
        }
        __syncthreads();
        return s;
    };

    // ---- pass 1: streaming row max / sum of exp ----
    float rm[8], rl[8];
#pragma unroll
    for (int r = 0; r < 8; ++r) { rm[r] = -3.0e38f; rl[r] = 0.f; }
    for (int j0 = 0; j0 < T_; j0 += 16) {
        v8f s = s_tile(j0);
#pragma unroll
        for (int r = 0; r < 8; ++r) {
            float mx = s[r];
#pragma unroll
            for (int off = 1; off < 16; off <<= 1) mx = fmaxf(mx, __shfl_xor(mx, off, 32));
            float nm = fmaxf(rm[r], mx);
            float e  = __expf(s[r] - nm);
#pragma unroll
            for (int off = 1; off < 16; off <<= 1) e += __shfl_xor(e, off, 32);
            rl[r] = rl[r] * __expf(rm[r] - nm) + e;
            rm[r] = nm;
        }
    }

    // ---- pass 2: normalized probs -> attn matrix mean + P x V ----
    v8f o0 = {}, o1 = {}, o2 = {}, o3 = {};
    for (int j0 = 0; j0 < T_; j0 += 32) {
#pragma unroll
        for (int sub = 0; sub < 2; ++sub) {
            v8f s = s_tile(j0 + sub * 16);
#pragma unroll
            for (int r = 0; r < 8; ++r) {
                const int m = r + half * 8;
                const int i = i0 + m;
                const int j = j0 + sub * 16 + l15;
                float p = (rl[r] > 0.f) ? __expf(s[r] - rm[r]) / rl[r] : 0.f;
                if (j > i + M_) p = 0.f;
                pbuf[m][sub * 16 + l15] = p;
                if (p != 0.f)
                    atomicAdd(&attnM[(size_t)i * T_ + j], p * (1.0f / (B_ * H_)));
            }
        }
        __syncthreads();
        v16bf pa;
#pragma unroll
        for (int e = 0; e < 8; ++e) {
            pa[e]     = (bf16_t)pbuf[l15][half * 16 + e];
            pa[e + 8] = (bf16_t)pbuf[l15][half * 16 + 8 + e];
        }
        const int kbase = j0 + half * 16;
        o0 = WMMA_BF16(pa, *(const v16bf*)(vB + (size_t)(0 * 16 + l15) * T_ + kbase), o0);
        o1 = WMMA_BF16(pa, *(const v16bf*)(vB + (size_t)(1 * 16 + l15) * T_ + kbase), o1);
        o2 = WMMA_BF16(pa, *(const v16bf*)(vB + (size_t)(2 * 16 + l15) * T_ + kbase), o2);
        o3 = WMMA_BF16(pa, *(const v16bf*)(vB + (size_t)(3 * 16 + l15) * T_ + kbase), o3);
        __syncthreads();
    }

#pragma unroll
    for (int r = 0; r < 8; ++r) {
        const int m = r + half * 8;
        const size_t i = (size_t)i0 + m;
        bf16_t* dst = avec + (i * B_ + b) * (size_t)(H_ * DH_) + h * DH_;
        dst[0  + l15] = (bf16_t)o0[r];
        dst[16 + l15] = (bf16_t)o1[r];
        dst[32 + l15] = (bf16_t)o2[r];
        dst[48 + l15] = (bf16_t)o3[r];
    }
}

// ---------------------------------------------------------------------------
// Residual + LayerNorm: out = LN(x + ao) * gamma + beta.   1 block / row.
// ---------------------------------------------------------------------------
__global__ __launch_bounds__(256) void ln_kernel(const float* __restrict__ x,
                                                 const float* __restrict__ ao,
                                                 const float* __restrict__ gamma,
                                                 const float* __restrict__ beta,
                                                 float* __restrict__ out) {
    __shared__ float ssum[256], ssq[256];
    const int row = blockIdx.x;
    const int tid = threadIdx.x;
    const float* xr = x  + (size_t)row * D_;
    const float* ar = ao + (size_t)row * D_;
    float s = 0.f, sq = 0.f, hv[4];
#pragma unroll
    for (int e = 0; e < 4; ++e) {
        int c = tid + e * 256;
        float v = xr[c] + ar[c];
        hv[e] = v; s += v; sq += v * v;
    }
    ssum[tid] = s; ssq[tid] = sq;
    __syncthreads();
    for (int off = 128; off > 0; off >>= 1) {
        if (tid < off) { ssum[tid] += ssum[tid + off]; ssq[tid] += ssq[tid + off]; }
        __syncthreads();
    }
    const float mean = ssum[0] * (1.0f / D_);
    const float var  = ssq[0] * (1.0f / D_) - mean * mean;
    const float inv  = rsqrtf(var + 1e-5f);
    float* orow = out + (size_t)row * D_;
#pragma unroll
    for (int e = 0; e < 4; ++e) {
        int c = tid + e * 256;
        orow[c] = (hv[e] - mean) * inv * gamma[c] + beta[c];
    }
}

// ---------------------------------------------------------------------------
// Launch
// ---------------------------------------------------------------------------
extern "C" void kernel_launch(void* const* d_in, const int* in_sizes, int n_in,
                              void* d_out, int out_size, void* d_ws, size_t ws_size,
                              hipStream_t stream) {
    (void)in_sizes; (void)n_in; (void)out_size; (void)ws_size;
    const float* x    = (const float*)d_in[0];
    const float* pos  = (const float*)d_in[1];
    const float* mem  = (const float*)d_in[2];
    const float* pbu  = (const float*)d_in[3];
    const float* pbv  = (const float*)d_in[4];
    /* d_in[5] = mask, recomputed analytically (j > i + M) */
    const float* Wq   = (const float*)d_in[6];
    const float* Wkv  = (const float*)d_in[7];
    const float* Wrel = (const float*)d_in[8];
    const float* Wo   = (const float*)d_in[9];
    const float* gam  = (const float*)d_in[10];
    const float* bet  = (const float*)d_in[11];

    float* out   = (float*)d_out;
    float* attnM = out + (size_t)L_ * B_ * D_;   // second output, [L, T]

    // workspace carve-out (~126 MB)
    char* w = (char*)d_ws;
    auto alloc = [&](size_t bytes) -> char* {
        char* p = w; w += (bytes + 255) & ~(size_t)255; return p;
    };
    bf16_t* c_bf   = (bf16_t*)alloc((size_t)T_ * B_ * D_ * 2);
    bf16_t* pos_bf = (bf16_t*)alloc((size_t)T_ * B_ * D_ * 2);
    bf16_t* WqT    = (bf16_t*)alloc((size_t)D_ * H_ * DH_ * 2);
    bf16_t* WkvT   = (bf16_t*)alloc((size_t)D_ * 2 * H_ * DH_ * 2);
    bf16_t* WrelT  = (bf16_t*)alloc((size_t)D_ * H_ * DH_ * 2);
    bf16_t* WoT    = (bf16_t*)alloc((size_t)H_ * DH_ * D_ * 2);
    float*  q_f    = (float*)alloc((size_t)L_ * B_ * H_ * DH_ * 4);
    float*  kv_f   = (float*)alloc((size_t)T_ * B_ * 2 * H_ * DH_ * 4);
    float*  r_f    = (float*)alloc((size_t)T_ * B_ * H_ * DH_ * 4);
    bf16_t* qu_bf  = (bf16_t*)alloc((size_t)B_ * H_ * L_ * DH_ * 2);
    bf16_t* qv_bf  = (bf16_t*)alloc((size_t)B_ * H_ * L_ * DH_ * 2);
    bf16_t* k_bf   = (bf16_t*)alloc((size_t)B_ * H_ * T_ * DH_ * 2);
    bf16_t* vT_bf  = (bf16_t*)alloc((size_t)B_ * H_ * DH_ * T_ * 2);
    bf16_t* r_bf   = (bf16_t*)alloc((size_t)B_ * H_ * T_ * DH_ * 2);
    bf16_t* av_bf  = (bf16_t*)alloc((size_t)L_ * B_ * H_ * DH_ * 2);
    float*  ao_f   = (float*)alloc((size_t)L_ * B_ * D_ * 4);

    const int TPB = 256;
    auto nb = [&](int n) { return (n + TPB - 1) / TPB; };

    zero_f32<<<nb(L_ * T_), TPB, 0, stream>>>(attnM, L_ * T_);

    // bf16 conversions; c = [memory ; x]
    cvt_f32_bf16<<<nb(M_ * B_ * D_), TPB, 0, stream>>>(mem, c_bf, M_ * B_ * D_);
    cvt_f32_bf16<<<nb(L_ * B_ * D_), TPB, 0, stream>>>(x, c_bf + (size_t)M_ * B_ * D_,
                                                       L_ * B_ * D_);
    cvt_f32_bf16<<<nb(T_ * B_ * D_), TPB, 0, stream>>>(pos, pos_bf, T_ * B_ * D_);
    cvt_f32_bf16_T<<<nb(D_ * H_ * DH_), TPB, 0, stream>>>(Wq, WqT, D_, H_ * DH_);
    cvt_f32_bf16_T<<<nb(D_ * 2 * H_ * DH_), TPB, 0, stream>>>(Wkv, WkvT, D_, 2 * H_ * DH_);
    cvt_f32_bf16_T<<<nb(D_ * H_ * DH_), TPB, 0, stream>>>(Wrel, WrelT, D_, H_ * DH_);
    cvt_f32_bf16_T<<<nb(H_ * DH_ * D_), TPB, 0, stream>>>(Wo, WoT, H_ * DH_, D_);

    // projection GEMMs (TDM-staged LDS panels, WMMA bf16, f32 accumulate)
    dim3 gq(L_ * B_ / 64, H_ * DH_ / 64);
    gemm_nt_bf16<<<gq, 128, 0, stream>>>(c_bf + (size_t)M_ * B_ * D_, WqT, q_f,
                                         L_ * B_, H_ * DH_, D_);
    dim3 gkv(T_ * B_ / 64, 2 * H_ * DH_ / 64);
    gemm_nt_bf16<<<gkv, 128, 0, stream>>>(c_bf, WkvT, kv_f, T_ * B_, 2 * H_ * DH_, D_);
    dim3 gr(T_ * B_ / 64, H_ * DH_ / 64);
    gemm_nt_bf16<<<gr, 128, 0, stream>>>(pos_bf, WrelT, r_f, T_ * B_, H_ * DH_, D_);

    // repack for attention layouts
    repack_qu<<<nb(B_ * H_ * L_ * DH_), TPB, 0, stream>>>(q_f, pbu, pbv, qu_bf, qv_bf);
    repack_kv<<<nb(B_ * H_ * T_ * DH_), TPB, 0, stream>>>(kv_f, k_bf, vT_bf);
    repack_r<<<nb(B_ * H_ * T_ * DH_), TPB, 0, stream>>>(r_f, r_bf);

    // streaming WMMA attention
    dim3 ga(L_ / 16, B_ * H_);
    attn_kernel<<<ga, 32, 0, stream>>>(qu_bf, qv_bf, k_bf, vT_bf, r_bf, attnM, av_bf);

    // output projection + residual layernorm
    dim3 go(L_ * B_ / 64, D_ / 64);
    gemm_nt_bf16<<<go, 128, 0, stream>>>(av_bf, WoT, ao_f, L_ * B_, D_, H_ * DH_);
    ln_kernel<<<L_ * B_, 256, 0, stream>>>(x, ao_f, gam, bet, out);
}
